// Video_TransfoXLModel_84988812853540
// MI455X (gfx1250) — matrix operand order repaired and checked by reading, not compiled, not run on hardware
//
#include <hip/hip_runtime.h>
#include <hip/hip_bf16.h>

// ---------------- model dims ----------------
#define QLEN   1024
#define BSZ    4
#define DIN    2048
#define DMODEL 1024
#define NHEAD  16
#define DHEAD  64
#define DINNER 4096
#define NTOK   (QLEN*BSZ)          // 4096 tokens, token = i*BSZ + b
#define NLAYER 4

typedef __attribute__((ext_vector_type(16))) __bf16 v16bf;
typedef __attribute__((ext_vector_type(8)))  float  v8f;
typedef unsigned long long ull;

// gfx1250 async global->LDS staging via inline CDNA5 asm (ISA 15.18.3 opcode 98).
#define USE_ASYNC_LDS 1

__device__ __forceinline__ unsigned short f2bf(float f) {
  unsigned int u = __float_as_uint(f);
  u += 0x7fffu + ((u >> 16) & 1u);     // round-to-nearest-even
  return (unsigned short)(u >> 16);
}

// ---------------- elementwise f32 -> bf16 cast ----------------
__global__ void __launch_bounds__(256) cast_bf16_k(const float* __restrict__ in,
                                                   unsigned short* __restrict__ out, int n) {
  int i = blockIdx.x * 256 + threadIdx.x;
  if (i < n) out[i] = f2bf(in[i]);
}

// ---------------- sinusoidal position embedding (bf16, [QLEN][DMODEL]) ----------------
__global__ void __launch_bounds__(256) pos_emb_k(unsigned short* __restrict__ rBF) {
  int idx = blockIdx.x * 256 + threadIdx.x;          // QLEN*DMODEL threads
  int j = idx >> 10;
  int m = idx & (DMODEL - 1);
  float pos  = (float)(QLEN - 1 - j);
  int   t    = m & 511;
  float freq = __expf(-((float)(2 * t) / (float)DMODEL) * __logf(10000.0f));
  float ang  = pos * freq;
  float v    = (m < 512) ? sinf(ang) : cosf(ang);
  rBF[idx] = f2bf(v);
}

// ---------------- bf16 WMMA GEMM: C = act(A[M,K] @ B[K,N] + bias) ----------------
// block = 256 threads (8 wave32). Block tile 128(M) x BN(N), K step 32, LDS double-buffered.
// BN=128: wave tile 32x64 (8 WMMA/step). BN=64: wave tile 32x32 (4 WMMA/step).
// grid = (N/BN, M/128, batch). Output: Cx[cxBase + z*strideCx + m*cxRow + n].
template<int BN>
__global__ void __launch_bounds__(256)
gemm_bf16_k(const unsigned short* __restrict__ A, int lda, ull strideA,
            const unsigned short* __restrict__ B, int ldb, ull strideB,
            float* __restrict__ Cf, ull cfBase, int cfRow, ull strideCf,
            unsigned short* __restrict__ Cb, ull cbBase, int cbRow, ull strideCb,
            const float* __restrict__ bias, int relu, int K)
{
  constexpr int NT = BN / 32;                  // 16-wide n-tiles per wave
  __shared__ unsigned short As[2][128 * 40];   // 128(M) x 32(K), padded stride 40
  __shared__ unsigned short Bs[2][BN * 40];    // BN(N) x 32(K), transposed, stride 40

  const int tid = threadIdx.x;
  const int z   = blockIdx.z;
  A += (ull)z * strideA;
  B += (ull)z * strideB;
  const ull cfB = cfBase + (ull)z * strideCf;
  const ull cbB = cbBase + (ull)z * strideCb;
  const int m0 = blockIdx.y * 128;
  const int n0 = blockIdx.x * BN;

  const int lane  = tid & 31;
  const int wave  = tid >> 5;
  const int wm    = (wave & 3) * 32;           // 4 waves along M
  const int wn    = (wave >> 2) * (NT * 16);   // 2 waves along N
  const int mlane = lane & 15;
  const int khalf = lane >> 4;
  const int kbA   = khalf * 8;                 // A frag: {kbA..kbA+7, kbA+16..kbA+23}
  const int kbB   = khalf * 8;                 // B frag: {kbB..kbB+7, kbB+16..kbB+23}

  // staging assignments
  const int arow = tid >> 2;                   // A: rows arow, arow+64; 8 K each
  const int akc  = (tid & 3) << 3;
  const int bnc  = (tid >> 4) * 8;             // B: 8 N values, 2 K rows
  const int bkp  = (tid & 15) * 2;

  auto stage = [&](int k0, int buf) {
    unsigned short* as = As[buf];
    unsigned short* bs = Bs[buf];
    const unsigned short* ga0 = A + (size_t)(m0 + arow) * lda + (k0 + akc);
    const unsigned short* ga1 = ga0 + (size_t)64 * lda;
#if USE_ASYNC_LDS
    {
      // VDST = per-lane LDS byte offset; VADDR = 64-bit global address (VFLAT GV mode).
      unsigned l0 = (unsigned)(unsigned long long)(const void*)&as[arow * 40 + akc];
      unsigned l1 = (unsigned)(unsigned long long)(const void*)&as[(arow + 64) * 40 + akc];
      asm volatile("global_load_async_to_lds_b128 %0, %1, off"
                   :: "v"(l0), "v"(ga0) : "memory");
      asm volatile("global_load_async_to_lds_b128 %0, %1, off"
                   :: "v"(l1), "v"(ga1) : "memory");
    }
#else
    *(uint4*)(&as[arow * 40 + akc])        = *(const uint4*)ga0;
    *(uint4*)(&as[(arow + 64) * 40 + akc]) = *(const uint4*)ga1;
#endif
    if (BN == 128 || tid < 128) {
      union { uint4 q; unsigned short s[8]; } r0, r1;
      const unsigned short* gb = B + (size_t)(k0 + bkp) * ldb + (n0 + bnc);
      r0.q = *(const uint4*)gb;
      r1.q = *(const uint4*)(gb + ldb);
      #pragma unroll
      for (int e = 0; e < 8; ++e) {
        unsigned int w = (unsigned int)r0.s[e] | ((unsigned int)r1.s[e] << 16);
        *(unsigned int*)(&bs[(bnc + e) * 40 + bkp]) = w;   // K pair packed as one dword
      }
    }
  };

  const v8f vz = {0.f,0.f,0.f,0.f,0.f,0.f,0.f,0.f};
  v8f acc[2][NT];
  #pragma unroll
  for (int i = 0; i < 2; ++i)
    #pragma unroll
    for (int j = 0; j < NT; ++j) acc[i][j] = vz;

  const int KT = K >> 5;
  stage(0, 0);
  int buf = 0;
  for (int kt = 0; kt < KT; ++kt) {
#if USE_ASYNC_LDS
    asm volatile("s_wait_asynccnt 0x0" ::: "memory");   // this wave's async stores done
#endif
    __syncthreads();                           // tile `buf` visible to all waves
    if (kt + 1 < KT) stage((kt + 1) << 5, buf ^ 1);
    if (kt + 2 < KT) {                         // speculative prefetch two tiles ahead
      __builtin_prefetch(A + (size_t)(m0 + arow) * lda + ((kt + 2) << 5) + akc, 0, 1);
      __builtin_prefetch(B + (size_t)(((kt + 2) << 5) + bkp) * ldb + (n0 + bnc), 0, 1);
    }

    const unsigned short* as = As[buf];
    const unsigned short* bs = Bs[buf];
    union FragU { uint4 u[2]; v16bf v; };
    FragU fa0, fa1;
    const unsigned short* p;
    p = &as[(wm + mlane) * 40 + kbA];      fa0.u[0] = *(const uint4*)p; fa0.u[1] = *(const uint4*)(p + 16);
    p = &as[(wm + 16 + mlane) * 40 + kbA]; fa1.u[0] = *(const uint4*)p; fa1.u[1] = *(const uint4*)(p + 16);
    #pragma unroll
    for (int t = 0; t < NT; ++t) {
      FragU fb;
      const unsigned short* q = &bs[(wn + t * 16 + mlane) * 40 + kbB];
      fb.u[0] = *(const uint4*)q; fb.u[1] = *(const uint4*)(q + 16);
      acc[0][t] = __builtin_amdgcn_wmma_f32_16x16x32_bf16(false, fa0.v, false, fb.v, (short)0, acc[0][t], false, false);
      acc[1][t] = __builtin_amdgcn_wmma_f32_16x16x32_bf16(false, fa1.v, false, fb.v, (short)0, acc[1][t], false, false);
    }
    buf ^= 1;
  }

  const int mrow = 8 * khalf;   // C/D layout: element r -> M = r + 8*(lane>>4), N = lane&15
  #pragma unroll
  for (int ti = 0; ti < 2; ++ti) {
    #pragma unroll
    for (int tj = 0; tj < NT; ++tj) {
      int   gn = n0 + wn + tj * 16 + mlane;
      float bv = bias ? bias[gn] : 0.0f;
      int   gmBase = m0 + wm + ti * 16 + mrow;
      #pragma unroll
      for (int r = 0; r < 8; ++r) {
        float v = acc[ti][tj][r] + bv;
        if (relu) v = fmaxf(v, 0.0f);
        size_t mm = (size_t)(gmBase + r);
        if (Cf) Cf[cfB + mm * (size_t)cfRow + gn] = v;
        if (Cb) Cb[cbB + mm * (size_t)cbRow + gn] = f2bf(v);
      }
    }
  }
}

// ---------------- qkv repack -> Qw/Qr [bn][i][d], KT [bn][d][j], V [bn][j][d] (bf16) ----------------
__global__ void __launch_bounds__(256)
repack_qkv_k(const float* __restrict__ qkv, const float* __restrict__ rwb, const float* __restrict__ rrb,
             unsigned short* __restrict__ Qw, unsigned short* __restrict__ Qr,
             unsigned short* __restrict__ KT, unsigned short* __restrict__ V)
{
  int idx   = blockIdx.x * 256 + threadIdx.x;   // NTOK*DMODEL threads
  int token = idx >> 10;
  int c     = idx & (DMODEL - 1);
  int i = token >> 2, b = token & 3;
  int n = c >> 6, d = c & 63;
  size_t base = (size_t)token * (3 * DMODEL);
  float q  = qkv[base + c];
  float kk = qkv[base + DMODEL + c];
  float vv = qkv[base + 2 * DMODEL + c];
  size_t zn = (size_t)(b * NHEAD + n);
  Qw[(zn * QLEN + i) * DHEAD + d] = f2bf(q + rwb[c]);
  Qr[(zn * QLEN + i) * DHEAD + d] = f2bf(q + rrb[c]);
  KT[(zn * DHEAD + d) * QLEN + i] = f2bf(kk);
  V [(zn * QLEN + i) * DHEAD + d] = f2bf(vv);
}

// ---------------- rk transpose: rk f32 [QLEN, DMODEL] -> RKT bf16 [DMODEL][QLEN] ----------------
__global__ void __launch_bounds__(256)
repack_rk_k(const float* __restrict__ rk, unsigned short* __restrict__ RKT) {
  int idx = blockIdx.x * 256 + threadIdx.x;     // QLEN*DMODEL
  int j = idx >> 10;
  int c = idx & (DMODEL - 1);
  RKT[(size_t)c * QLEN + j] = f2bf(rk[idx]);
}

// ---------------- rel-shift + softmax: prob[z,i,j] = softmax_j((AC + relshift(BD)) * scale) ----------------
__global__ void __launch_bounds__(256)
softmax_relshift_k(const float* __restrict__ AC, const float* __restrict__ BD,
                   unsigned short* __restrict__ prob, float scale)
{
  const int i = blockIdx.x;            // 0..QLEN-1
  const int z = blockIdx.y;            // head 0..15 (within one batch element)
  const int tid = threadIdx.x;
  const float* ac = AC + ((size_t)z * QLEN + i) * QLEN;
  const float* bd = BD + (size_t)z * QLEN * QLEN;
  unsigned short* pr = prob + ((size_t)z * QLEN + i) * QLEN;

  __shared__ float red[256];
  float v[4];
  float mx = -3.4e38f;
  #pragma unroll
  for (int t = 0; t < 4; ++t) {
    int j   = tid + (t << 8);
    int idx = (i + 1) * QLEN + j;                 // TXL rel_shift index math (q = k = QLEN)
    int row = idx / (QLEN + 1);
    int col = idx % (QLEN + 1);
    float bdv = col ? bd[(size_t)row * QLEN + (col - 1)] : 0.0f;
    v[t] = (ac[j] + bdv) * scale;
    mx = fmaxf(mx, v[t]);
  }
  red[tid] = mx; __syncthreads();
  for (int s = 128; s > 0; s >>= 1) { if (tid < s) red[tid] = fmaxf(red[tid], red[tid + s]); __syncthreads(); }
  mx = red[0]; __syncthreads();

  float sum = 0.0f;
  #pragma unroll
  for (int t = 0; t < 4; ++t) { v[t] = expf(v[t] - mx); sum += v[t]; }
  red[tid] = sum; __syncthreads();
  for (int s = 128; s > 0; s >>= 1) { if (tid < s) red[tid] += red[tid + s]; __syncthreads(); }
  float inv = 1.0f / red[0];
  #pragma unroll
  for (int t = 0; t < 4; ++t) pr[tid + (t << 8)] = f2bf(v[t] * inv);
}

// ---------------- residual + LayerNorm: h = LN(h + add)*g + b ; writes f32 + bf16 ----------------
__global__ void __launch_bounds__(256)
add_ln_k(const float* __restrict__ h, const float* __restrict__ add,
         const float* __restrict__ g, const float* __restrict__ bta,
         float* __restrict__ houtF, unsigned short* __restrict__ houtB)
{
  const int row = blockIdx.x;       // token 0..NTOK-1
  const int tid = threadIdx.x;
  const float* hr = h   + (size_t)row * DMODEL;
  const float* ar = add + (size_t)row * DMODEL;

  __shared__ float red[256];
  float y[4]; float s = 0.0f, s2 = 0.0f;
  #pragma unroll
  for (int t = 0; t < 4; ++t) {
    int c = tid + (t << 8);
    y[t] = hr[c] + ar[c];
    s += y[t]; s2 += y[t] * y[t];
  }
  red[tid] = s; __syncthreads();
  for (int st = 128; st > 0; st >>= 1) { if (tid < st) red[tid] += red[tid + st]; __syncthreads(); }
  float mean = red[0] * (1.0f / DMODEL); __syncthreads();
  red[tid] = s2; __syncthreads();
  for (int st = 128; st > 0; st >>= 1) { if (tid < st) red[tid] += red[tid + st]; __syncthreads(); }
  float var = red[0] * (1.0f / DMODEL) - mean * mean;
  float rstd = rsqrtf(var + 1e-5f);
  #pragma unroll
  for (int t = 0; t < 4; ++t) {
    int c = tid + (t << 8);
    float o = (y[t] - mean) * rstd * g[c] + bta[c];
    houtF[(size_t)row * DMODEL + c] = o;
    houtB[(size_t)row * DMODEL + c] = f2bf(o);
  }
}

__global__ void __launch_bounds__(256)
copy_f32_k(const float* __restrict__ in, float* __restrict__ out, int n) {
  int i = blockIdx.x * 256 + threadIdx.x;
  if (i < n) out[i] = in[i];
}

// ==================== host orchestration ====================
extern "C" void kernel_launch(void* const* d_in, const int* in_sizes, int n_in,
                              void* d_out, int out_size, void* d_ws, size_t ws_size,
                              hipStream_t stream)
{
  (void)in_sizes; (void)n_in; (void)out_size; (void)ws_size;
  const float* x    = (const float*)d_in[0];
  const float* Win  = (const float*)d_in[1];
  const float* Wqkv = (const float*)d_in[2];
  const float* Wr   = (const float*)d_in[3];
  const float* Wo   = (const float*)d_in[4];
  const float* ln1g = (const float*)d_in[5];
  const float* ln1b = (const float*)d_in[6];
  const float* W1   = (const float*)d_in[7];
  const float* b1   = (const float*)d_in[8];
  const float* W2   = (const float*)d_in[9];
  const float* b2   = (const float*)d_in[10];
  const float* ln2g = (const float*)d_in[11];
  const float* ln2b = (const float*)d_in[12];
  const float* rwb  = (const float*)d_in[13];
  const float* rrb  = (const float*)d_in[14];

  char* ws = (char*)d_ws;
  size_t off = 0;
  auto alloc = [&](size_t bytes) -> char* {
    char* p = ws + off;
    off = (off + bytes + 255) & ~(size_t)255;
    return p;
  };
  typedef unsigned short ush;

  ush*  xBF     = (ush*) alloc((size_t)NTOK * DIN * 2);
  ush*  WinBF   = (ush*) alloc((size_t)DIN * DMODEL * 2);
  ush*  WqkvBF  = (ush*) alloc((size_t)NLAYER * DMODEL * 3 * DMODEL * 2);
  ush*  WrBF    = (ush*) alloc((size_t)NLAYER * DMODEL * DMODEL * 2);
  ush*  WoBF    = (ush*) alloc((size_t)NLAYER * DMODEL * DMODEL * 2);
  ush*  W1BF    = (ush*) alloc((size_t)NLAYER * DMODEL * DINNER * 2);
  ush*  W2BF    = (ush*) alloc((size_t)NLAYER * DINNER * DMODEL * 2);
  ush*  rBF     = (ush*) alloc((size_t)QLEN * DMODEL * 2);
  float* hF32   = (float*)alloc((size_t)NTOK * DMODEL * 4);
  ush*  hBF     = (ush*) alloc((size_t)NTOK * DMODEL * 2);
  float* qkvF32 = (float*)alloc((size_t)NTOK * 3 * DMODEL * 4);
  float* rkF32  = (float*)alloc((size_t)QLEN * DMODEL * 4);
  ush*  QwBF    = (ush*) alloc((size_t)BSZ * NHEAD * QLEN * DHEAD * 2);
  ush*  QrBF    = (ush*) alloc((size_t)BSZ * NHEAD * QLEN * DHEAD * 2);
  ush*  KTBF    = (ush*) alloc((size_t)BSZ * NHEAD * DHEAD * QLEN * 2);
  ush*  VBF     = (ush*) alloc((size_t)BSZ * NHEAD * QLEN * DHEAD * 2);
  ush*  RKTBF   = (ush*) alloc((size_t)DMODEL * QLEN * 2);
  float* ACb    = (float*)alloc((size_t)NHEAD * QLEN * QLEN * 4);   // reused per batch element
  float* BDb    = (float*)alloc((size_t)NHEAD * QLEN * QLEN * 4);
  ush*  probB   = (ush*) alloc((size_t)NHEAD * QLEN * QLEN * 2);
  ush*  avBF    = (ush*) alloc((size_t)NTOK * DMODEL * 2);
  float* tmpF32 = (float*)alloc((size_t)NTOK * DMODEL * 4);
  ush*  ff1BF   = (ush*) alloc((size_t)NTOK * DINNER * 2);

  auto cast = [&](const float* src, ush* dst, long long n) {
    cast_bf16_k<<<dim3((unsigned)((n + 255) / 256)), dim3(256), 0, stream>>>(src, dst, (int)n);
  };
  auto gemm = [&](const ush* A, int lda, ull sA,
                  const ush* B, int ldb, ull sB,
                  float* Cf, ull cfBase, int cfRow, ull sCf,
                  ush* Cb, ull cbBase, int cbRow, ull sCb,
                  const float* bias, int relu, int M, int N, int K, int batch) {
    if ((N & 127) == 0) {
      dim3 grid((unsigned)(N / 128), (unsigned)(M / 128), (unsigned)batch);
      gemm_bf16_k<128><<<grid, dim3(256), 0, stream>>>(A, lda, sA, B, ldb, sB,
          Cf, cfBase, cfRow, sCf, Cb, cbBase, cbRow, sCb, bias, relu, K);
    } else {
      dim3 grid((unsigned)(N / 64), (unsigned)(M / 128), (unsigned)batch);
      gemm_bf16_k<64><<<grid, dim3(256), 0, stream>>>(A, lda, sA, B, ldb, sB,
          Cf, cfBase, cfRow, sCf, Cb, cbBase, cbRow, sCb, bias, relu, K);
    }
  };

  // --- weight / input casts (done every call; deterministic) ---
  cast(x,    xBF,    (long long)NTOK * DIN);
  cast(Win,  WinBF,  (long long)DIN * DMODEL);
  cast(Wqkv, WqkvBF, (long long)NLAYER * DMODEL * 3 * DMODEL);
  cast(Wr,   WrBF,   (long long)NLAYER * DMODEL * DMODEL);
  cast(Wo,   WoBF,   (long long)NLAYER * DMODEL * DMODEL);
  cast(W1,   W1BF,   (long long)NLAYER * DMODEL * DINNER);
  cast(W2,   W2BF,   (long long)NLAYER * DINNER * DMODEL);

  pos_emb_k<<<dim3((QLEN * DMODEL) / 256), dim3(256), 0, stream>>>(rBF);

  // --- input projection: h = x @ Win ---
  gemm(xBF, DIN, 0, WinBF, DMODEL, 0,
       hF32, 0, DMODEL, 0, hBF, 0, DMODEL, 0,
       nullptr, 0, NTOK, DMODEL, DIN, 1);

  const float scale = 1.0f / 8.0f;   // 1/sqrt(64)
  const ull HS = (ull)QLEN * DHEAD;  // 65536 per (b,n)
  const ull SS = (ull)QLEN * QLEN;   // score plane

  for (int l = 0; l < NLAYER; ++l) {
    const ush* Wqkv_l = WqkvBF + (size_t)l * DMODEL * 3 * DMODEL;
    const ush* Wr_l   = WrBF   + (size_t)l * DMODEL * DMODEL;
    const ush* Wo_l   = WoBF   + (size_t)l * DMODEL * DMODEL;
    const ush* W1_l   = W1BF   + (size_t)l * DMODEL * DINNER;
    const ush* W2_l   = W2BF   + (size_t)l * DINNER * DMODEL;
    const float* b1_l = b1 + (size_t)l * DINNER;
    const float* b2_l = b2 + (size_t)l * DMODEL;

    // qkv = h @ Wqkv[l]
    gemm(hBF, DMODEL, 0, Wqkv_l, 3 * DMODEL, 0,
         qkvF32, 0, 3 * DMODEL, 0, nullptr, 0, 0, 0,
         nullptr, 0, NTOK, 3 * DMODEL, DMODEL, 1);
    // rk = r @ Wr[l]
    gemm(rBF, DMODEL, 0, Wr_l, DMODEL, 0,
         rkF32, 0, DMODEL, 0, nullptr, 0, 0, 0,
         nullptr, 0, QLEN, DMODEL, DMODEL, 1);

    repack_qkv_k<<<dim3((NTOK * DMODEL) / 256), dim3(256), 0, stream>>>(
        qkvF32, rwb, rrb, QwBF, QrBF, KTBF, VBF);
    repack_rk_k<<<dim3((QLEN * DMODEL) / 256), dim3(256), 0, stream>>>(rkF32, RKTBF);

    for (int b = 0; b < BSZ; ++b) {
      const ush* Qw_b = QwBF + (size_t)b * NHEAD * HS;
      const ush* Qr_b = QrBF + (size_t)b * NHEAD * HS;
      const ush* KT_b = KTBF + (size_t)b * NHEAD * HS;
      const ush* V_b  = VBF  + (size_t)b * NHEAD * HS;

      // AC[n] = Qw[b,n] @ KT[b,n]   (batch = 16 heads)
      gemm(Qw_b, DHEAD, HS, KT_b, QLEN, HS,
           ACb, 0, QLEN, SS, nullptr, 0, 0, 0,
           nullptr, 0, QLEN, QLEN, DHEAD, NHEAD);
      // BD[n] = Qr[b,n] @ RKT[n]
      gemm(Qr_b, DHEAD, HS, RKTBF, QLEN, HS,
           BDb, 0, QLEN, SS, nullptr, 0, 0, 0,
           nullptr, 0, QLEN, QLEN, DHEAD, NHEAD);
      // prob = softmax((AC + relshift(BD)) * scale)
      softmax_relshift_k<<<dim3(QLEN, NHEAD), dim3(256), 0, stream>>>(ACb, BDb, probB, scale);
      // attn_vec[b,n] = prob @ V  -> scattered into token-major [NTOK, DMODEL] bf16
      gemm(probB, QLEN, SS, V_b, DHEAD, HS,
           nullptr, 0, 0, 0,
           avBF, (ull)b * DMODEL, BSZ * DMODEL, (ull)DHEAD,
           nullptr, 0, QLEN, DHEAD, QLEN, NHEAD);
    }

    // attn_out = attn_vec @ Wo[l]
    gemm(avBF, DMODEL, 0, Wo_l, DMODEL, 0,
         tmpF32, 0, DMODEL, 0, nullptr, 0, 0, 0,
         nullptr, 0, NTOK, DMODEL, DMODEL, 1);
    // h = LN(h + attn_out)
    add_ln_k<<<dim3(NTOK), dim3(256), 0, stream>>>(hF32, tmpF32,
        ln1g + (size_t)l * DMODEL, ln1b + (size_t)l * DMODEL, hF32, hBF);

    // ff1 = relu(h @ W1 + b1) (bf16 only)
    gemm(hBF, DMODEL, 0, W1_l, DINNER, 0,
         nullptr, 0, 0, 0, ff1BF, 0, DINNER, 0,
         b1_l, 1, NTOK, DINNER, DMODEL, 1);
    // ff2 = ff1 @ W2 + b2
    gemm(ff1BF, DINNER, 0, W2_l, DMODEL, 0,
         tmpF32, 0, DMODEL, 0, nullptr, 0, 0, 0,
         b2_l, 0, NTOK, DMODEL, DINNER, 1);
    // h = LN(h + ff2)
    add_ln_k<<<dim3(NTOK), dim3(256), 0, stream>>>(hF32, tmpF32,
        ln2g + (size_t)l * DMODEL, ln2b + (size_t)l * DMODEL, hF32, hBF);
  }

  copy_f32_k<<<dim3((NTOK * DMODEL) / 256), dim3(256), 0, stream>>>(
      hF32, (float*)d_out, NTOK * DMODEL);
}
